// Decoder_40415642255407
// MI455X (gfx1250) — matrix operand order repaired and checked by reading, not compiled, not run on hardware
//
#include <hip/hip_runtime.h>
#include <hip/hip_bf16.h>
#include <stdint.h>

typedef __attribute__((ext_vector_type(16))) _Float16 v16h;
typedef __attribute__((ext_vector_type(8)))  float    v8f;

#define B_    8
#define F0_   64
#define FIN_  192
#define FH_   64
#define FOUT_ 128
#define LAT_  512
#define P_    45376
#define I0_   12288   // end of w0 (64x192)
#define I1_   12352   // end of b0
#define I2_   20544   // end of w1 (128x64)
#define I3_   20672   // end of b1
#define I4_   45248   // end of ws (128x192); bs = [I4_, P_)
#define LRELU_ 0.2f
#define NT_   32      // pixels per block in fused kernel
#define TILE_HALFS_ (FIN_*NT_)   // 6144 halfs = 12KB per (b,tile)

__device__ __forceinline__ float lrelu(float x){ return x > 0.f ? x : LRELU_*x; }

// ---- WMMA operand loaders per CDNA5 ISA 7.12.2 (wave32) -------------------
// A: 16x32 f16 (MxK). lane L: row = L&15, hg = L>>4.
//    half h -> K = h + 16*hg   (16 contiguous halfs, matches B)
__device__ __forceinline__ v16h ldA16(const _Float16* base, int stride){
  int lane = threadIdx.x & 31, row = lane & 15, hg = lane >> 4;
  v16h a;
#pragma unroll
  for (int h = 0; h < 16; ++h){
    int k = h + 16*hg;
    a[h] = base[row*stride + k];
  }
  return a;
}
// B: 32x16 f16 (KxN). lane L: col = L&15, hg = L>>4; half h -> K = h + 16*hg
// (K-contiguous per lane; same k order as ldA16)
// Swizzled LDS tile: block `blk` holds 32 lanes x 16 halfs contiguously.
__device__ __forceinline__ v16h ldB_sw(const _Float16* buf, int blk){
  int lane = threadIdx.x & 31;
  const _Float16* src = buf + (blk*32 + lane)*16;   // 32B contiguous per lane
  v16h b;
#pragma unroll
  for (int h = 0; h < 16; ++h) b[h] = src[h];
  return b;
}
__device__ __forceinline__ v16h ldB_sw_lrelu(const _Float16* buf, int blk){
  int lane = threadIdx.x & 31;
  const _Float16* src = buf + (blk*32 + lane)*16;
  v16h b;
#pragma unroll
  for (int h = 0; h < 16; ++h){
    float v = (float)src[h];
    b[h] = (_Float16)lrelu(v);
  }
  return b;
}
// B tile where memory is N-major (rows = N, stride over K), f16 source:
// per lane 16 contiguous halfs (32B) -> two global_load_b128
__device__ __forceinline__ v16h ldB_nmajor_f16(const _Float16* base, int strideK){
  int lane = threadIdx.x & 31, col = lane & 15, hg = lane >> 4;
  const _Float16* src = base + (size_t)col*strideK + 16*hg;
  v16h b;
#pragma unroll
  for (int h = 0; h < 16; ++h) b[h] = src[h];
  return b;
}
#define WMMA_F32F16(a,b,c) __builtin_amdgcn_wmma_f32_16x16x32_f16(false,(a),false,(b),(short)0,(c),false,false)

// ---- CDNA5 async global->LDS copy (ASYNCcnt-tracked, no VGPR roundtrip) ---
__device__ __forceinline__ void async_ld_b128(uint32_t lds_byte_off, uint64_t gaddr){
  asm volatile("global_load_async_to_lds_b128 %0, %1, off"
               :: "v"(lds_byte_off), "v"(gaddr) : "memory");
}
__device__ __forceinline__ void wait_async0(){
  asm volatile("s_wait_asynccnt 0x0" ::: "memory");
}

// swizzled p address (halfs) within a batch-level plane of numTiles tiles
__device__ __forceinline__ size_t p_swz(int numTiles, int b, int ch, int pix){
  int tile = pix >> 5, j = pix & 31, nt = j >> 4, col = j & 15;
  int kk = ch >> 5, kr = ch & 31, hg = kr >> 4, h = kr & 15;
  return ((size_t)b*numTiles + tile)*TILE_HALFS_
       + (size_t)(((kk*2 + nt)*32 + (col + 16*hg))*16 + h);
}

// ---- deterministic gaussian noise (stand-in for jax PRNG) -----------------
__device__ __forceinline__ uint32_t pcg(uint32_t v){
  v = v*747796405u + 2891336453u;
  uint32_t w = ((v >> ((v >> 28) + 4u)) ^ v) * 277803737u;
  return (w >> 22) ^ w;
}
__device__ __forceinline__ float nrand(uint32_t s){
  uint32_t a = pcg(s), b = pcg(s ^ 0x9E3779B9u);
  float u1 = (float)(a & 0xFFFFFFu) * (1.0f/16777216.0f) + 1e-7f;
  float u2 = (float)(b & 0xFFFFFFu) * (1.0f/16777216.0f);
  return sqrtf(-2.0f*logf(u1)) * cosf(6.28318530718f*u2);
}

// ===========================================================================
__global__ void k_init_out(const float* __restrict__ seed, float* __restrict__ out){
  int idx = blockIdx.x*blockDim.x + threadIdx.x;
  if (idx < B_*F0_*256) out[idx] = seed[idx % (F0_*256)];
}

__global__ void k_pack_static(const float* __restrict__ convs_w, const float* __restrict__ conv0_w,
                              const float* __restrict__ conv1_w,
                              _Float16* __restrict__ convs_h, _Float16* __restrict__ conv0_h,
                              _Float16* __restrict__ conv1_h){
  int idx = blockIdx.x*blockDim.x + threadIdx.x;
  if (idx < FOUT_*FIN_) convs_h[idx] = (_Float16)convs_w[idx];
  if (idx < FH_*FIN_)   conv0_h[idx] = (_Float16)conv0_w[idx];
  if (idx < FOUT_*FH_)  conv1_h[idx] = (_Float16)conv1_w[idx];
}

// one-time f32 -> f16 conversion of dyna_w (iteration-invariant; read 16x/launch)
__global__ void k_cvt_dynaw(const float* __restrict__ src, _Float16* __restrict__ dst){
  size_t idx = ((size_t)blockIdx.x*blockDim.x + threadIdx.x)*4u;
  if (idx + 3 < (size_t)P_*LAT_){
    float4 v = *(const float4*)(src + idx);
    dst[idx+0] = (_Float16)v.x; dst[idx+1] = (_Float16)v.y;
    dst[idx+2] = (_Float16)v.z; dst[idx+3] = (_Float16)v.w;
  }
}

// Sobel features + identity, instance-norm; p stored f16 in fragment-swizzled
// tile order so the fused kernel can async-copy one contiguous 12KB block and
// build every WMMA B fragment from 32 contiguous LDS bytes per lane.
__global__ void k_prep(const float* __restrict__ out, _Float16* __restrict__ p,
                       int H, int W, int numTiles){
  int b  = blockIdx.x / FIN_;
  int pc = blockIdx.x % FIN_;
  int sc = pc % F0_, mode = pc / F0_;
  int HWl = H*W;
  const float* src = out + (size_t)(b*F0_ + sc)*HWl;
  float s = 0.f, s2 = 0.f;
  for (int i = threadIdx.x; i < HWl; i += blockDim.x){
    int y = i / W, x = i % W;
    float v;
    if (mode == 2) v = src[i];
    else {
      auto at = [&](int yy, int xx) -> float {
        return (yy < 0 || yy >= H || xx < 0 || xx >= W) ? 0.f : src[yy*W + xx];
      };
      if (mode == 0)
        v = (-at(y-1,x-1) + at(y-1,x+1) - 2.f*at(y,x-1) + 2.f*at(y,x+1)
             - at(y+1,x-1) + at(y+1,x+1)) * 0.125f;
      else
        v = (-at(y-1,x-1) - 2.f*at(y-1,x) - at(y-1,x+1)
             + at(y+1,x-1) + 2.f*at(y+1,x) + at(y+1,x+1)) * 0.125f;
    }
    p[p_swz(numTiles, b, pc, i)] = (_Float16)v;
    s += v; s2 += v*v;
  }
  __shared__ float rs[256], rq[256];
  rs[threadIdx.x] = s; rq[threadIdx.x] = s2;
  __syncthreads();
  for (int o = 128; o > 0; o >>= 1){
    if ((int)threadIdx.x < o){ rs[threadIdx.x] += rs[threadIdx.x+o]; rq[threadIdx.x] += rq[threadIdx.x+o]; }
    __syncthreads();
  }
  float m   = rs[0] / (float)HWl;
  float var = rq[0] / (float)HWl - m*m;
  float inv = rsqrtf(fmaxf(var, 0.f) + 1e-5f);
  for (int i = threadIdx.x; i < HWl; i += blockDim.x){
    size_t a = p_swz(numTiles, b, pc, i);
    p[a] = (_Float16)(((float)p[a] - m) * inv);
  }
}

// latc[b,o] = lat[b,:] . lat_exp_w[c][o,:] + lat_exp_b[c][o]; padded 16 rows, f16
__global__ void k_latc(const float* __restrict__ lat, const float* __restrict__ lew,
                       const float* __restrict__ leb, _Float16* __restrict__ latc_h, int c){
  int idx = blockIdx.x*blockDim.x + threadIdx.x;
  if (idx >= 16*LAT_) return;
  int b = idx / LAT_, o = idx % LAT_;
  float v = 0.f;
  if (b < B_){
    const float* wrow = lew + ((size_t)c*LAT_ + o)*LAT_;
    const float* lrow = lat + (size_t)b*LAT_;
    for (int i = 0; i < LAT_; ++i) v += lrow[i]*wrow[i];
    v += leb[c*LAT_ + o];
  }
  latc_h[idx] = (_Float16)v;
}

// ks = latc @ dyna_w^T + dyna_b  (M=16 pad of 8, N=45376, K=512)  -- WMMA
__global__ void k_ks(const _Float16* __restrict__ latc_h, const _Float16* __restrict__ dyna_wh,
                     const float* __restrict__ dyna_b, float* __restrict__ ks){
  __shared__ _Float16 Atile[16*LAT_];
  for (int i = threadIdx.x; i < 16*LAT_; i += blockDim.x) Atile[i] = latc_h[i];
  __syncthreads();
  int wave = threadIdx.x >> 5;
  int n0 = blockIdx.x*64 + wave*16;
  v8f acc = {};
  for (int kk = 0; kk < LAT_/32; ++kk){
    int k0 = kk*32;
    if (kk + 1 < LAT_/32)
      __builtin_prefetch(dyna_wh + (size_t)n0*LAT_ + k0 + 32, 0, 1);
    v16h a = ldA16(Atile + k0, LAT_);
    v16h b = ldB_nmajor_f16(dyna_wh + (size_t)n0*LAT_ + k0, LAT_);
    acc = WMMA_F32F16(a, b, acc);
  }
  int lane = threadIdx.x & 31;
  int n = n0 + (lane & 15);
  int mh = (lane >> 4)*8;
#pragma unroll
  for (int r = 0; r < 8; ++r){
    int m = r + mh;
    if (m < B_) ks[(size_t)m*P_ + n] = acc[r] + dyna_b[n];
  }
}

// slice ks -> f16 dynamic weight matrices (biases stay f32 inside ks)
__global__ void k_pack_dyn(const float* __restrict__ ks, _Float16* __restrict__ w0h,
                           _Float16* __restrict__ w1h, _Float16* __restrict__ wsh){
  int idx = blockIdx.x*blockDim.x + threadIdx.x;
  if (idx >= B_*P_) return;
  int b = idx / P_, p = idx % P_;
  float v = ks[idx];
  if      (p < I0_)               w0h[(size_t)b*(FH_*FIN_)   + p]        = (_Float16)v;
  else if (p >= I1_ && p < I2_)   w1h[(size_t)b*(FOUT_*FH_)  + (p-I1_)]  = (_Float16)v;
  else if (p >= I3_ && p < I4_)   wsh[(size_t)b*(FOUT_*FIN_) + (p-I3_)]  = (_Float16)v;
}

// fused per-pixel chain: async-staged swizzled p tile, 3 WMMA GEMM stages,
// GLU + noise + integrate.  block = 256 threads (8 waves), 32-pixel tile.
__global__ void k_fused(const _Float16* __restrict__ p, float* __restrict__ out,
                        const _Float16* __restrict__ convs_h, const _Float16* __restrict__ conv0_h,
                        const _Float16* __restrict__ conv1_h,
                        const _Float16* __restrict__ w0h, const _Float16* __restrict__ w1h,
                        const _Float16* __restrict__ wsh,
                        const float* __restrict__ ks, const float* __restrict__ conv0_b,
                        const float* __restrict__ conv1_b, const float* __restrict__ noise_w,
                        const float* __restrict__ leak_factor, int c, int HWl)
{
  __shared__ _Float16 ph [TILE_HALFS_];  // swizzled p tile (12 blocks)   12KB
  __shared__ _Float16 dxh[FOUT_*NT_];    // swizzled stage-2 tile (8 blk)  8KB
  __shared__ float acc1[256*NT_];        // [xs ; x_s]                    32KB
  __shared__ float acc2[256*NT_];        // [dx1 ; dxd1]                  32KB

  int b    = blockIdx.y;
  int tile = blockIdx.x;
  int pix0 = tile * NT_;
  int numTiles = HWl / NT_;

  // ---- async global->LDS stage: one contiguous 12KB block ------------------
  {
    uint64_t gbase = (uint64_t)(uintptr_t)(p + ((size_t)b*numTiles + tile)*TILE_HALFS_);
    uint32_t lbase = (uint32_t)(uintptr_t)(&ph[0]);
#pragma unroll
    for (int it = 0; it < 3; ++it){
      int slot = (int)threadIdx.x + 256*it;      // 768 x 16B transfers
      async_ld_b128(lbase + (uint32_t)slot*16u, gbase + (uint64_t)slot*16u);
    }
    wait_async0();
  }
  __syncthreads();

  int wave = threadIdx.x >> 5;
  int lane = threadIdx.x & 31;
  int col  = lane & 15, mh = (lane >> 4)*8;

  // ---- GEMM1: M=256 rows [convs_w ; ws_b], K=192, B = ph -------------------
  for (int t = 0; t < 4; ++t){
    int id = wave*4 + t;          // 0..31
    int mt = id >> 1, nt = id & 1;
    const _Float16* A = (mt < 8) ? (convs_h + mt*16*FIN_)
                                 : (wsh + (size_t)b*FOUT_*FIN_ + (mt-8)*16*FIN_);
    v8f acc = {};
    for (int kk = 0; kk < FIN_/32; ++kk){
      v16h av = ldA16(A + kk*32, FIN_);
      v16h bv = ldB_sw(ph, kk*2 + nt);
      acc = WMMA_F32F16(av, bv, acc);
    }
#pragma unroll
    for (int r = 0; r < 8; ++r) acc1[(mt*16 + r + mh)*NT_ + nt*16 + col] = acc[r];
  }

  // ---- GEMM2: M=128 rows [conv0_w ; w0_b], K=192, B = lrelu(ph) -> dxh -----
  const float* b0 = ks + (size_t)b*P_ + I0_;
  for (int t = 0; t < 2; ++t){
    int id = wave*2 + t;          // 0..15
    int mt = id >> 1, nt = id & 1;
    const _Float16* A = (mt < 4) ? (conv0_h + mt*16*FIN_)
                                 : (w0h + (size_t)b*FH_*FIN_ + (mt-4)*16*FIN_);
    v8f acc = {};
    for (int kk = 0; kk < FIN_/32; ++kk){
      v16h av = ldA16(A + kk*32, FIN_);
      v16h bv = ldB_sw_lrelu(ph, kk*2 + nt);
      acc = WMMA_F32F16(av, bv, acc);
    }
    // write swizzled dxh: element (m, nt*16+col) -> block ((m>>5)*2+nt),
    // lane' = col + 16*((m>>4)&1), half = m&15
#pragma unroll
    for (int r = 0; r < 8; ++r){
      int m = mt*16 + r + mh;
      float bias = (m < FH_) ? conv0_b[m] : b0[m - FH_];
      int blk = ((m >> 5) << 1) + nt;
      int lp  = col + (((m >> 4) & 1) << 4);
      dxh[(blk*32 + lp)*16 + (m & 15)] = (_Float16)lrelu(acc[r] + bias);
    }
  }
  __syncthreads();

  // ---- GEMM3: dx1 = conv1 x dxh[K 0:64], dxd1 = w1_b x dxh[K 64:128] -------
  for (int t = 0; t < 4; ++t){
    int id = wave*4 + t;          // 0..31
    int mt = id >> 1, nt = id & 1;
    const _Float16* A;
    int koff;                      // K-origin into dxh (channels)
    if (mt < 8){ A = conv1_h + mt*16*FH_;                        koff = 0;  }
    else       { A = w1h + (size_t)b*FOUT_*FH_ + (mt-8)*16*FH_;  koff = 64; }
    v8f acc = {};
    for (int kk = 0; kk < FH_/32; ++kk){
      v16h av = ldA16(A + kk*32, FH_);
      v16h bv = ldB_sw(dxh, ((koff >> 5) + kk)*2 + nt);
      acc = WMMA_F32F16(av, bv, acc);
    }
#pragma unroll
    for (int r = 0; r < 8; ++r) acc2[(mt*16 + r + mh)*NT_ + nt*16 + col] = acc[r];
  }
  __syncthreads();

  // ---- epilogue: o = f + d, GLU, noise, out += lf*o ------------------------
  const float* b1 = ks + (size_t)b*P_ + I2_;
  const float* bs = ks + (size_t)b*P_ + I4_;
  float lf = fminf(fmaxf(leak_factor[0], 0.001f), 1000.f);
  for (int i = threadIdx.x; i < F0_*NT_; i += blockDim.x){
    int ch = i >> 5, j = i & 31;
    int chg = ch + F0_;
    float oa = acc1[ch*NT_+j]  + acc1[(128+ch)*NT_+j]  + bs[ch]
             + 0.1f*(acc2[ch*NT_+j]  + conv1_b[ch])
             + 0.1f*(acc2[(128+ch)*NT_+j]  + b1[ch]);
    float og = acc1[chg*NT_+j] + acc1[(128+chg)*NT_+j] + bs[chg]
             + 0.1f*(acc2[chg*NT_+j] + conv1_b[chg])
             + 0.1f*(acc2[(128+chg)*NT_+j] + b1[chg]);
    float val = oa * (1.f/(1.f + expf(-og)));
    int pixg = pix0 + j;
    float nz = nrand(((uint32_t)c*0x9E3779B9u) ^ ((uint32_t)b*0x85EBCA77u) ^ (uint32_t)pixg);
    val += noise_w[c*F0_ + ch]*nz;
    out[(size_t)(b*F0_ + ch)*HWl + pixg] += lf*val;
  }
}

// bilinear x2 (half-pixel centers, edge clamp)
__global__ void k_upsample(const float* __restrict__ src, float* __restrict__ dst, int H, int W){
  int H2 = 2*H, W2 = 2*W;
  int total = B_*F0_*H2*W2;
  int idx = blockIdx.x*blockDim.x + threadIdx.x;
  if (idx >= total) return;
  int x = idx % W2; int y = (idx / W2) % H2; int bc = idx / (W2*H2);
  float sy = (y + 0.5f)*0.5f - 0.5f;
  float sx = (x + 0.5f)*0.5f - 0.5f;
  int y0 = (int)floorf(sy), x0 = (int)floorf(sx);
  float ty = sy - (float)y0, tx = sx - (float)x0;
  int y0c = min(max(y0,   0), H-1), y1c = min(max(y0+1, 0), H-1);
  int x0c = min(max(x0,   0), W-1), x1c = min(max(x0+1, 0), W-1);
  const float* s = src + (size_t)bc*H*W;
  float v = (1.f-ty)*((1.f-tx)*s[y0c*W+x0c] + tx*s[y0c*W+x1c])
          +      ty *((1.f-tx)*s[y1c*W+x0c] + tx*s[y1c*W+x1c]);
  dst[idx] = v;
}

// gaussian 3x3 depthwise, zero padding
__global__ void k_blur(const float* __restrict__ src, float* __restrict__ dst, int H, int W){
  const float GA = 0.6065306597126334f;
  const float NORM = 1.f/((1.f + 2.f*GA)*(1.f + 2.f*GA));
  int total = B_*F0_*H*W;
  int idx = blockIdx.x*blockDim.x + threadIdx.x;
  if (idx >= total) return;
  int x = idx % W; int y = (idx / W) % H; int bc = idx / (W*H);
  const float* s = src + (size_t)bc*H*W;
  auto at = [&](int yy, int xx) -> float {
    return (yy < 0 || yy >= H || xx < 0 || xx >= W) ? 0.f : s[yy*W + xx];
  };
  float v = at(y-1,x-1)*GA*GA + at(y-1,x)*GA + at(y-1,x+1)*GA*GA
          + at(y  ,x-1)*GA    + at(y  ,x)    + at(y  ,x+1)*GA
          + at(y+1,x-1)*GA*GA + at(y+1,x)*GA + at(y+1,x+1)*GA*GA;
  dst[idx] = v*NORM;
}

// 3-channel head: clip + raw, concatenated in d_out
__global__ void k_head(const float* __restrict__ out, const float* __restrict__ out_w,
                       const float* __restrict__ out_b, float* __restrict__ d_out, int HWl){
  int idx = blockIdx.x*blockDim.x + threadIdx.x;
  if (idx >= B_*HWl) return;
  int b = idx / HWl, pix = idx % HWl;
  const float* o = out + (size_t)b*F0_*HWl + pix;
#pragma unroll
  for (int k = 0; k < 3; ++k){
    float v = out_b[k];
    for (int ch = 0; ch < F0_; ++ch) v += out_w[k*F0_ + ch]*o[(size_t)ch*HWl];
    size_t oi = (size_t)(b*3 + k)*HWl + pix;
    d_out[oi] = fminf(fmaxf(v, -1.f), 1.f);
    d_out[(size_t)B_*3*HWl + oi] = v;
  }
}

// ===========================================================================
extern "C" void kernel_launch(void* const* d_in, const int* in_sizes, int n_in,
                              void* d_out, int out_size, void* d_ws, size_t ws_size,
                              hipStream_t stream){
  (void)in_sizes; (void)n_in; (void)out_size; (void)ws_size;
  const float* lat      = (const float*)d_in[0];
  const float* seed     = (const float*)d_in[1];
  const float* leak     = (const float*)d_in[2];
  const float* lew      = (const float*)d_in[3];
  const float* leb      = (const float*)d_in[4];
  const float* noise_w  = (const float*)d_in[5];
  const float* dyna_w   = (const float*)d_in[6];
  const float* dyna_b   = (const float*)d_in[7];
  const float* conv0_w  = (const float*)d_in[8];
  const float* conv0_b  = (const float*)d_in[9];
  const float* conv1_w  = (const float*)d_in[10];
  const float* conv1_b  = (const float*)d_in[11];
  const float* convs_w  = (const float*)d_in[12];
  const float* out_w    = (const float*)d_in[13];
  const float* out_b    = (const float*)d_in[14];

  char* ws = (char*)d_ws;
  size_t off = 0;
  auto carve = [&](size_t bytes) -> char* {
    char* pt = ws + off;
    off = (off + bytes + 255) & ~(size_t)255;
    return pt;
  };
  float*     outbuf   = (float*)    carve((size_t)B_*F0_ *128*128*4);
  float*     tmpbuf   = (float*)    carve((size_t)B_*F0_ *128*128*4);
  _Float16*  pbuf     = (_Float16*) carve((size_t)B_*FIN_*128*128*2);
  float*     ksbuf    = (float*)    carve((size_t)B_*P_*4);
  _Float16*  dyna_wh  = (_Float16*) carve((size_t)P_*LAT_*2);
  _Float16*  latc_h   = (_Float16*) carve((size_t)16*LAT_*2);
  _Float16*  convs_h  = (_Float16*) carve((size_t)FOUT_*FIN_*2);
  _Float16*  conv0_h  = (_Float16*) carve((size_t)FH_ *FIN_*2);
  _Float16*  conv1_h  = (_Float16*) carve((size_t)FOUT_*FH_ *2);
  _Float16*  w0h      = (_Float16*) carve((size_t)B_*FH_ *FIN_*2);
  _Float16*  w1h      = (_Float16*) carve((size_t)B_*FOUT_*FH_ *2);
  _Float16*  wsh      = (_Float16*) carve((size_t)B_*FOUT_*FIN_*2);

  k_init_out<<<dim3((B_*F0_*256 + 255)/256), dim3(256), 0, stream>>>(seed, outbuf);
  k_pack_static<<<dim3((FOUT_*FIN_ + 255)/256), dim3(256), 0, stream>>>(
      convs_w, conv0_w, conv1_w, convs_h, conv0_h, conv1_h);
  {
    size_t n4 = ((size_t)P_*LAT_ + 3)/4;
    k_cvt_dynaw<<<dim3((unsigned)((n4 + 255)/256)), dim3(256), 0, stream>>>(dyna_w, dyna_wh);
  }

  int H = 16;
  for (int c = 0; c < 16; ++c){
    int W = H, HWl = H*W, numTiles = HWl / NT_;
    k_prep<<<dim3(B_*FIN_), dim3(256), 0, stream>>>(outbuf, pbuf, H, W, numTiles);
    k_latc<<<dim3((16*LAT_ + 255)/256), dim3(256), 0, stream>>>(lat, lew, leb, latc_h, c);
    k_ks<<<dim3(P_/64), dim3(128), 0, stream>>>(latc_h, dyna_wh, dyna_b, ksbuf);
    k_pack_dyn<<<dim3((B_*P_ + 255)/256), dim3(256), 0, stream>>>(ksbuf, w0h, w1h, wsh);
    k_fused<<<dim3(numTiles, B_), dim3(256), 0, stream>>>(
        pbuf, outbuf, convs_h, conv0_h, conv1_h, w0h, w1h, wsh,
        ksbuf, conv0_b, conv1_b, noise_w, leak, c, HWl);
    if (c < 15 && (c % 4) == 3){
      int H2 = 2*H;
      k_upsample<<<dim3((B_*F0_*H2*H2 + 255)/256), dim3(256), 0, stream>>>(outbuf, tmpbuf, H, W);
      k_blur<<<dim3((B_*F0_*H2*H2 + 255)/256), dim3(256), 0, stream>>>(tmpbuf, outbuf, H2, H2);
      H = H2;
    }
  }
  int HWf = H*H;
  k_head<<<dim3((B_*HWf + 255)/256), dim3(256), 0, stream>>>(outbuf, out_w, out_b, (float*)d_out, HWf);
}
